// RangeRet_73340861547101
// MI455X (gfx1250) — matrix operand order repaired and checked by reading, not compiled, not run on hardware
//
#include <hip/hip_runtime.h>
#include <hip/hip_bf16.h>
#include <math.h>

// ---------------- model dims ----------------
#define BATCH   64
#define LSEQ    512
#define HID     128
#define FFND    256
#define NHEADS  4
#define DHEAD   32
#define VHEAD   64
#define VDIM    256
#define NROWS   (BATCH * LSEQ)      // 32768
#define EPSLN   1e-5f

typedef __attribute__((ext_vector_type(16))) _Float16 v16h;
typedef __attribute__((ext_vector_type(8)))  float    v8f;

// ---------------- helpers ----------------
__device__ __forceinline__ float gelu_exact(float v) {
    return 0.5f * v * (1.0f + erff(v * 0.70710678118654752f));
}
__device__ __forceinline__ float wave_red_sum(float v) {
    #pragma unroll
    for (int o = 16; o > 0; o >>= 1) v += __shfl_xor(v, o, 32);
    return v;
}
__device__ __forceinline__ float head_gamma(int h) {
    const float lg0 = -3.4657359028f;   // log(1/32)
    const float lg1 = -6.2383246250f;   // log(1/512)
    return 1.0f - __expf(lg0 + (lg1 - lg0) * (float)h * (1.0f / 3.0f));
}

// ---------------- embed MLP: gelu(gelu(gelu(x@w1)@w2)@w3) ----------------
__global__ __launch_bounds__(128)
void embed_kernel(const float* __restrict__ x,
                  const float* __restrict__ w1, const float* __restrict__ b1,
                  const float* __restrict__ w2, const float* __restrict__ b2,
                  const float* __restrict__ w3, const float* __restrict__ b3,
                  float* __restrict__ X) {
    int row = blockIdx.x;
    int t   = threadIdx.x;
    __shared__ float xr[5], h1[32], h2[64];
    if (t < 5) xr[t] = x[row * 5 + t];
    __syncthreads();
    if (t < 32) {
        float s = b1[t];
        #pragma unroll
        for (int i = 0; i < 5; ++i) s += xr[i] * w1[i * 32 + t];
        h1[t] = gelu_exact(s);
    }
    __syncthreads();
    if (t < 64) {
        float s = b2[t];
        #pragma unroll
        for (int i = 0; i < 32; ++i) s += h1[i] * w2[i * 64 + t];
        h2[t] = gelu_exact(s);
    }
    __syncthreads();
    float s = b3[t];
    #pragma unroll
    for (int i = 0; i < 64; ++i) s += h2[i] * w3[i * 128 + t];
    X[(size_t)row * HID + t] = gelu_exact(s);
}

// ---------------- layernorm: one wave per 128-wide row ----------------
__global__ __launch_bounds__(256)
void ln_kernel(const float* __restrict__ in, const float* __restrict__ w,
               const float* __restrict__ b, float* __restrict__ out) {
    int wave = threadIdx.x >> 5;
    int lane = threadIdx.x & 31;
    int row  = blockIdx.x * 8 + wave;
    const float* r = in + (size_t)row * HID;
    float v[4];
    float s = 0.f, sq = 0.f;
    #pragma unroll
    for (int j = 0; j < 4; ++j) {
        v[j] = r[lane * 4 + j];
        s += v[j]; sq += v[j] * v[j];
    }
    s  = wave_red_sum(s);
    sq = wave_red_sum(sq);
    float mu  = s * (1.0f / HID);
    float var = sq * (1.0f / HID) - mu * mu;
    float rn  = rsqrtf(var + EPSLN);
    float* o = out + (size_t)row * HID;
    #pragma unroll
    for (int j = 0; j < 4; ++j) {
        int idx = lane * 4 + j;
        o[idx] = (v[j] - mu) * rn * w[idx] + b[idx];
    }
}

// ---------------- QKV projection + xpos, outputs f16 ----------------
// Q,K layout: [b*4+n][l][32]; V layout TRANSPOSED: [b*4+n][v][LSEQ]
__global__ __launch_bounds__(256)
void qkv_kernel(const float* __restrict__ Xn,
                const float* __restrict__ wq, const float* __restrict__ wk,
                const float* __restrict__ wv,
                _Float16* __restrict__ Qh, _Float16* __restrict__ Kh,
                _Float16* __restrict__ Vh) {
    int row = blockIdx.x;
    int b   = row / LSEQ;
    int l   = row % LSEQ;
    int t   = threadIdx.x;
    __shared__ float xr[HID];
    __shared__ float qk[256];   // raw q(0..127), k(128..255)
    if (t < HID) xr[t] = Xn[(size_t)row * HID + t];
    __syncthreads();

    // raw Q or K value for output slot t
    {
        int isK = (t >= 128);
        int o   = t & 127;
        int n   = o >> 5;
        int d   = o & 31;
        const float* wb = (isK ? wk : wq) + ((size_t)n * HID) * DHEAD + d;
        float s = 0.f;
        #pragma unroll 8
        for (int h = 0; h < HID; ++h) s += xr[h] * wb[h * DHEAD];
        qk[t] = s;
    }
    // V output slot t (256 outputs), stored transposed [bn][vv][l]
    {
        int n  = t >> 6;
        int vv = t & 63;
        const float* wb = wv + ((size_t)n * HID) * VHEAD + vv;
        float s = 0.f;
        #pragma unroll 8
        for (int h = 0; h < HID; ++h) s += xr[h] * wb[h * VHEAD];
        Vh[((size_t)(b * NHEADS + n) * VHEAD + vv) * LSEQ + l] = (_Float16)s;
    }
    __syncthreads();

    // xpos rotation
    {
        int isK = (t >= 128);
        int o   = t & 127;
        int n   = o >> 5;
        int d   = o & 31;
        int i   = d >> 1;
        float base = ((float)i + 12.8f) / 44.8f;
        float p    = (float)l * (1.0f / 512.0f);
        float sc   = powf(base, isK ? -p : p);
        float invf = powf(10000.0f, -(float)i / 16.0f);
        float ang  = (float)l * invf;
        float c = cosf(ang) * sc;
        float s = sinf(ang) * sc;
        float me = qk[t];
        float pr = (d & 1) ? qk[t - 1] : qk[t + 1];
        float outv = (d & 1) ? (me * c + pr * s) : (me * c - pr * s);
        _Float16* dst = isK ? Kh : Qh;
        dst[(((size_t)(b * NHEADS + n)) * LSEQ + l) * DHEAD + d] = (_Float16)outv;
    }
}

// ---------------- retention core: WMMA flash-style ----------------
// grid.x = BATCH*NHEADS*(LSEQ/128); block = 128 (4 waves, 32 rows each)
__global__ __launch_bounds__(128)
void retention_kernel(const _Float16* __restrict__ Q,
                      const _Float16* __restrict__ K,
                      const _Float16* __restrict__ V,
                      float* __restrict__ Y) {
    const int blk    = blockIdx.x;
    const int rowblk = blk & 3;          // LSEQ/128 = 4 slabs
    const int bn     = blk >> 2;         // b*4 + head
    const int head   = bn & 3;
    const int bIdx   = bn >> 2;
    const int tid    = threadIdx.x;
    const int wave   = tid >> 5;
    const int lane   = tid & 31;
    const int g      = lane >> 4;
    const int hl     = lane & 15;
    const int R      = rowblk * 128;
    const int rbw    = R + wave * 32;    // this wave's 32-row base

    const _Float16* Qh = Q + (size_t)bn * LSEQ * DHEAD;
    const _Float16* Kh = K + (size_t)bn * LSEQ * DHEAD;
    const _Float16* Vh = V + (size_t)bn * VHEAD * LSEQ;   // transposed [v][l]

    const float gamma = head_gamma(head);
    const float lgam  = logf(gamma);

    // hoisted decay factors: gamma^(l-m) = basef(chunk,rI) * rowf[vv] * colf * adj^tI
    const float colf = __expf(-lgam * (float)hl);     // gamma^(-hl)
    const float adj  = __expf(-lgam * 16.0f);         // gamma^(-16)
    float rowf[8];
    #pragma unroll
    for (int vv = 0; vv < 8; ++vv) rowf[vv] = __expf(lgam * (float)(g * 8 + vv));

    __shared__ _Float16 Kt[32][32];           // K rows of this chunk
    __shared__ _Float16 Vt[64][32];           // V transposed: [v][m_rel]
    __shared__ _Float16 Pt[4][2][16][32];     // decayed scores per wave / row-subtile

    // A-operands: this wave's two 16x32 Q row tiles (ISA A-layout gather).
    v16h a_q[2];
    #pragma unroll
    for (int rI = 0; rI < 2; ++rI) {
        const _Float16* qrow = Qh + (size_t)(rbw + rI * 16 + hl) * DHEAD;
        #pragma unroll
        for (int p = 0; p < 8; ++p) {
            int k0 = (p < 4) ? (p * 2 + g * 8) : (16 + (p - 4) * 2 + g * 8);
            a_q[rI][2 * p]     = qrow[k0];
            a_q[rI][2 * p + 1] = qrow[k0 + 1];
        }
    }

    const v8f zf = {0.f, 0.f, 0.f, 0.f, 0.f, 0.f, 0.f, 0.f};
    v8f acc[2][4];
    #pragma unroll
    for (int rI = 0; rI < 2; ++rI)
        #pragma unroll
        for (int j = 0; j < 4; ++j) acc[rI][j] = zf;

    const int nchunks = rowblk * 4 + 4;       // m0 up to R+127 in steps of 32
    for (int c = 0; c < nchunks; ++c) {
        const int m0 = c * 32;
        __syncthreads();
        // stage K chunk (32x32 halves) as flat dword copy
        {
            const unsigned int* src = (const unsigned int*)(Kh + (size_t)m0 * DHEAD);
            unsigned int* dst = (unsigned int*)&Kt[0][0];
            for (int i = tid; i < 512; i += 128) dst[i] = src[i];
        }
        // stage V chunk: global already transposed, rows of 32 halves
        {
            unsigned int* dst = (unsigned int*)&Vt[0][0];
            for (int i = tid; i < 1024; i += 128) {
                int vv = i >> 4, mc = i & 15;
                dst[i] = *((const unsigned int*)(Vh + (size_t)vv * LSEQ + m0) + mc);
            }
        }
        __syncthreads();

        // per-chunk decay base factors (one exp per row subtile)
        float basef[2];
        #pragma unroll
        for (int rI = 0; rI < 2; ++rI)
            basef[rI] = __expf(lgam * (float)(rbw + rI * 16 - m0));

        // scores for two 16-column tiles x two row subtiles
        #pragma unroll
        for (int tI = 0; tI < 2; ++tI) {
            v16h b_k;
            #pragma unroll
            for (int v = 0; v < 8; ++v) {
                int kk = (v < 4) ? (v * 2 + g * 8) : (16 + (v - 4) * 2 + g * 8);
                b_k[2 * v]     = Kt[tI * 16 + hl][kk];
                b_k[2 * v + 1] = Kt[tI * 16 + hl][kk + 1];
            }
            #pragma unroll
            for (int rI = 0; rI < 2; ++rI) {
                v8f s = __builtin_amdgcn_wmma_f32_16x16x32_f16(
                    false, a_q[rI], false, b_k, (short)0, zf, false, false);
                float fb   = basef[rI] * colf * (tI ? adj : 1.0f);
                int  dbase = (rbw + rI * 16 - m0) + g * 8 - tI * 16 - hl; // l-m at vv=0
                #pragma unroll
                for (int vv = 0; vv < 8; ++vv) {
                    float f = ((dbase + vv) >= 0) ? fb * rowf[vv] : 0.f;
                    Pt[wave][rI][g * 8 + vv][tI * 16 + hl] = (_Float16)(s[vv] * f);
                }
            }
        }
        // wave-private Pt: LDS ops are in-order within a wave, no block barrier needed

        v16h a_p[2];
        #pragma unroll
        for (int rI = 0; rI < 2; ++rI) {
            #pragma unroll
            for (int p = 0; p < 8; ++p) {
                int k0 = (p < 4) ? (p * 2 + g * 8) : (16 + (p - 4) * 2 + g * 8);
                a_p[rI][2 * p]     = Pt[wave][rI][hl][k0];
                a_p[rI][2 * p + 1] = Pt[wave][rI][hl][k0 + 1];
            }
        }
        // 4 output column groups of 16 over v=64, B operand reused across row subtiles
        #pragma unroll
        for (int j = 0; j < 4; ++j) {
            v16h b_v;
            #pragma unroll
            for (int v = 0; v < 8; ++v) {
                int kk = (v < 4) ? (v * 2 + g * 8) : (16 + (v - 4) * 2 + g * 8);
                b_v[2 * v]     = Vt[j * 16 + hl][kk];
                b_v[2 * v + 1] = Vt[j * 16 + hl][kk + 1];
            }
            #pragma unroll
            for (int rI = 0; rI < 2; ++rI) {
                acc[rI][j] = __builtin_amdgcn_wmma_f32_16x16x32_f16(
                    false, a_p[rI], false, b_v, (short)0, acc[rI][j], false, false);
            }
        }
    }

    // write Y[b, l, head*64 + j*16 + N]  (already in (B,L,256) layout)
    #pragma unroll
    for (int rI = 0; rI < 2; ++rI) {
        #pragma unroll
        for (int j = 0; j < 4; ++j) {
            #pragma unroll
            for (int vv = 0; vv < 8; ++vv) {
                int l = rbw + rI * 16 + g * 8 + vv;
                Y[((size_t)(bIdx * LSEQ + l)) * VDIM + head * VHEAD + j * 16 + hl] =
                    acc[rI][j][vv];
            }
        }
    }
}

// ---------------- group-norm + swish gate + wo + residual ----------------
__global__ __launch_bounds__(128)
void gate_kernel(const float* __restrict__ Y, const float* __restrict__ Xn,
                 const float* __restrict__ Xres,
                 const float* __restrict__ wg, const float* __restrict__ wo,
                 const float* __restrict__ gnw, const float* __restrict__ gnb,
                 float* __restrict__ out) {
    int row  = blockIdx.x;
    int t    = threadIdx.x;
    int wave = t >> 5;
    int lane = t & 31;
    __shared__ float yn[VDIM];
    __shared__ float xr[HID];
    __shared__ float tt[VDIM];

    // per-head groupnorm (head = wave)
    {
        const float* yr = Y + (size_t)row * VDIM + wave * VHEAD;
        float y0 = yr[lane], y1 = yr[lane + 32];
        float s  = wave_red_sum(y0 + y1);
        float sq = wave_red_sum(y0 * y0 + y1 * y1);
        float mu  = s * (1.0f / VHEAD);
        float var = sq * (1.0f / VHEAD) - mu * mu;
        float rn  = rsqrtf(var + EPSLN);
        int i0 = wave * VHEAD + lane, i1 = i0 + 32;
        yn[i0] = (y0 - mu) * rn * gnw[i0] + gnb[i0];
        yn[i1] = (y1 - mu) * rn * gnw[i1] + gnb[i1];
    }
    if (t < HID) xr[t] = Xn[(size_t)row * HID + t];
    __syncthreads();

    // G = Xn@wg, T = silu(G)*Yn  (2 outputs per thread)
    #pragma unroll
    for (int rep = 0; rep < 2; ++rep) {
        int o = t + rep * 128;
        float gv = 0.f;
        #pragma unroll 8
        for (int h = 0; h < HID; ++h) gv += xr[h] * wg[h * VDIM + o];
        tt[o] = gv / (1.0f + __expf(-gv)) * yn[o];
    }
    __syncthreads();

    // out = T@wo + Xres
    float s = 0.f;
    #pragma unroll 8
    for (int cI = 0; cI < VDIM; ++cI) s += tt[cI] * wo[cI * HID + t];
    out[(size_t)row * HID + t] = s + Xres[(size_t)row * HID + t];
}

// ---------------- FFN: gelu(Z@w1+b1)@w2 + b2 + residual ----------------
__global__ __launch_bounds__(256)
void ffn_kernel(const float* __restrict__ Z, const float* __restrict__ w1,
                const float* __restrict__ b1, const float* __restrict__ w2,
                const float* __restrict__ b2, const float* __restrict__ Yres,
                float* __restrict__ out) {
    int row = blockIdx.x;
    int t   = threadIdx.x;
    __shared__ float zr[HID];
    __shared__ float hh[FFND];
    if (t < HID) zr[t] = Z[(size_t)row * HID + t];
    __syncthreads();
    {
        float s = b1[t];
        #pragma unroll 8
        for (int h = 0; h < HID; ++h) s += zr[h] * w1[h * FFND + t];
        hh[t] = gelu_exact(s);
    }
    __syncthreads();
    if (t < HID) {
        float s = b2[t];
        #pragma unroll 8
        for (int cI = 0; cI < FFND; ++cI) s += hh[cI] * w2[cI * HID + t];
        out[(size_t)row * HID + t] = s + Yres[(size_t)row * HID + t];
    }
}

// ---------------- decoder + softmax ----------------
__global__ __launch_bounds__(64)
void dec_kernel(const float* __restrict__ X,
                const float* __restrict__ w1, const float* __restrict__ b1,
                const float* __restrict__ w2, const float* __restrict__ b2,
                float* __restrict__ out) {
    int row = blockIdx.x;
    int t   = threadIdx.x;
    __shared__ float xr[HID];
    __shared__ float hh[64];
    __shared__ float lg[20];
    xr[t]      = X[(size_t)row * HID + t];
    xr[t + 64] = X[(size_t)row * HID + t + 64];
    __syncthreads();
    {
        float s = b1[t];
        #pragma unroll 8
        for (int h = 0; h < HID; ++h) s += xr[h] * w1[h * 64 + t];
        hh[t] = gelu_exact(s);
    }
    __syncthreads();
    if (t < 20) {
        float s = b2[t];
        #pragma unroll
        for (int h = 0; h < 64; ++h) s += hh[h] * w2[h * 20 + t];
        lg[t] = s;
    }
    __syncthreads();
    if (t < 20) {
        float mx = lg[0];
        #pragma unroll
        for (int i = 1; i < 20; ++i) mx = fmaxf(mx, lg[i]);
        float denom = 0.f;
        #pragma unroll
        for (int i = 0; i < 20; ++i) denom += __expf(lg[i] - mx);
        out[(size_t)row * 20 + t] = __expf(lg[t] - mx) / denom;
    }
}

// ---------------- host orchestration ----------------
extern "C" void kernel_launch(void* const* d_in, const int* in_sizes, int n_in,
                              void* d_out, int out_size, void* d_ws, size_t ws_size,
                              hipStream_t stream) {
    const float* x      = (const float*)d_in[0];
    const float* rem_w1 = (const float*)d_in[1];
    const float* rem_b1 = (const float*)d_in[2];
    const float* rem_w2 = (const float*)d_in[3];
    const float* rem_b2 = (const float*)d_in[4];
    const float* rem_w3 = (const float*)d_in[5];
    const float* rem_b3 = (const float*)d_in[6];
    const float* wq     = (const float*)d_in[7];
    const float* wk     = (const float*)d_in[8];
    const float* wv     = (const float*)d_in[9];
    const float* wg     = (const float*)d_in[10];
    const float* wo     = (const float*)d_in[11];
    const float* gn_w   = (const float*)d_in[12];
    const float* gn_b   = (const float*)d_in[13];
    const float* ln1_w  = (const float*)d_in[14];
    const float* ln1_b  = (const float*)d_in[15];
    const float* ln2_w  = (const float*)d_in[16];
    const float* ln2_b  = (const float*)d_in[17];
    const float* ffn_w1 = (const float*)d_in[18];
    const float* ffn_b1 = (const float*)d_in[19];
    const float* ffn_w2 = (const float*)d_in[20];
    const float* ffn_b2 = (const float*)d_in[21];
    const float* dec_w1 = (const float*)d_in[22];
    const float* dec_b1 = (const float*)d_in[23];
    const float* dec_w2 = (const float*)d_in[24];
    const float* dec_b2 = (const float*)d_in[25];

    const size_t MB = 1024ull * 1024ull;
    char* ws = (char*)d_ws;
    float*    X   = (float*)(ws + 0 * MB);      // 16 MB activations
    float*    Bf1 = (float*)(ws + 16 * MB);     // 16 MB LN output
    float*    Bf2 = (float*)(ws + 32 * MB);     // 16 MB MSR residual
    _Float16* Qh  = (_Float16*)(ws + 48 * MB);  // 8 MB
    _Float16* Kh  = (_Float16*)(ws + 56 * MB);  // 8 MB
    _Float16* Vh  = (_Float16*)(ws + 64 * MB);  // 16 MB (transposed layout)
    float*    Y2  = (float*)(ws + 80 * MB);     // 32 MB retention out
    (void)ws_size; (void)n_in; (void)in_sizes; (void)out_size;

    embed_kernel<<<NROWS, 128, 0, stream>>>(x, rem_w1, rem_b1, rem_w2, rem_b2,
                                            rem_w3, rem_b3, X);

    for (int l = 0; l < 4; ++l) {
        const float* wql = wq + (size_t)l * NHEADS * HID * DHEAD;
        const float* wkl = wk + (size_t)l * NHEADS * HID * DHEAD;
        const float* wvl = wv + (size_t)l * NHEADS * HID * VHEAD;
        const float* wgl = wg + (size_t)l * HID * VDIM;
        const float* wol = wo + (size_t)l * VDIM * HID;

        ln_kernel<<<NROWS / 8, 256, 0, stream>>>(X, ln1_w + l * HID, ln1_b + l * HID, Bf1);
        qkv_kernel<<<NROWS, 256, 0, stream>>>(Bf1, wql, wkl, wvl, Qh, Kh, Vh);
        retention_kernel<<<BATCH * NHEADS * (LSEQ / 128), 128, 0, stream>>>(Qh, Kh, Vh, Y2);
        gate_kernel<<<NROWS, 128, 0, stream>>>(Y2, Bf1, X, wgl, wol,
                                               gn_w + l * VDIM, gn_b + l * VDIM, Bf2);
        ln_kernel<<<NROWS / 8, 256, 0, stream>>>(Bf2, ln2_w + l * HID, ln2_b + l * HID, Bf1);
        ffn_kernel<<<NROWS, 256, 0, stream>>>(Bf1, ffn_w1 + (size_t)l * HID * FFND,
                                              ffn_b1 + l * FFND,
                                              ffn_w2 + (size_t)l * FFND * HID,
                                              ffn_b2 + l * HID, Bf2, X);
    }

    dec_kernel<<<NROWS, 64, 0, stream>>>(X, dec_w1, dec_b1, dec_w2, dec_b2, (float*)d_out);
}
